// SwinTransformerUnet_76828374991581
// MI455X (gfx1250) — matrix operand order repaired and checked by reading, not compiled, not run on hardware
//
#include <hip/hip_runtime.h>

typedef __bf16 bf16;
typedef bf16  v16bf __attribute__((ext_vector_type(16)));
typedef float v8f   __attribute__((ext_vector_type(8)));

#define TOKENS   393216ull   // B*H*W = 8*128*384
#define CDIM     192
#define C3       576
#define MLPH     768
#define NWIN     768         // windows per image
#define NHEAD    6
#define HDIM     32
#define NTOK     64          // tokens per window
#define SCALE_F  0.17677669529663689f  // 32^-0.5

// ---------- helpers ----------
__device__ __forceinline__ bf16 f2bf(float f) {
  unsigned u = __float_as_uint(f);
  unsigned r = (u + 0x7FFFu + ((u >> 16) & 1u)) >> 16;
  unsigned short s = (unsigned short)r;
  return __builtin_bit_cast(bf16, s);
}

__device__ __forceinline__ v8f wmma_bf16(v16bf a, v16bf b, v8f c) {
  return __builtin_amdgcn_wmma_f32_16x16x32_bf16(false, a, false, b, (short)0, c, false, false);
}

// async 16-byte global->LDS copy (GVS addressing: sgpr base + per-lane byte offset)
__device__ __forceinline__ void async_ld_b128(unsigned lds_addr, unsigned voff,
                                              unsigned long long gbase) {
  asm volatile("global_load_async_to_lds_b128 %0, %1, %2"
               :: "v"(lds_addr), "v"(voff), "s"(gbase) : "memory");
}
__device__ __forceinline__ void wait_async0() {
  asm volatile("s_wait_asynccnt 0x0" ::: "memory");
}

// A fragment 16x32 (rows row0..row0+15, k k0..k0+31) from row-major bf16 [*, ld]
__device__ __forceinline__ v16bf load_a_frag(const bf16* s, int ld, int row0, int k0, int lane) {
  union { v16bf v; bf16 e[16]; } u;
  const bf16* p = s + (size_t)(row0 + (lane & 15)) * ld + k0 + ((lane >> 4) << 4);
  #pragma unroll
  for (int i = 0; i < 16; ++i) u.e[i] = p[i];
  return u.v;
}

// B fragment 32x16, element (k,n) stored n-major: s[n*ld + k]  (weight rows / K rows)
__device__ __forceinline__ v16bf load_b_frag_nmaj(const bf16* s, int ld, int k0, int n0, int lane) {
  union { v16bf v; bf16 e[16]; } u;
  const bf16* p = s + (size_t)(n0 + (lane & 15)) * ld + k0 + ((lane >> 4) << 4);
  #pragma unroll
  for (int i = 0; i < 16; ++i) u.e[i] = p[i];
  return u.v;
}

// B fragment 32x16, element (k,n) stored k-major: s[k*ld + n]
__device__ __forceinline__ v16bf load_b_frag_kmaj(const bf16* s, int ld, int k0, int n0, int lane) {
  union { v16bf v; bf16 e[16]; } u;
  const bf16* p = s + (size_t)(k0 + ((lane >> 4) << 4)) * ld + n0 + (lane & 15);
  #pragma unroll
  for (int i = 0; i < 16; ++i) u.e[i] = p[(size_t)i * ld];
  return u.v;
}

// ---------- generic 64x64 tile mainloop (A row-major [M,K], W row-major [N,K]) ----------
// block = 128 threads (4 waves); wave w computes rows [w*16, w*16+16) x all 64 cols.
// Tile staging uses async global->LDS b128 copies (ASYNCcnt) instead of VGPR bounce.
__device__ __forceinline__ void mm_mainloop(const bf16* __restrict__ A,
                                            const bf16* __restrict__ Wt, int K,
                                            size_t row0, int col0,
                                            bf16* As, bf16* Bs, v8f acc[4],
                                            int tid, int lane, int wave) {
  v8f z = {};
  #pragma unroll
  for (int t = 0; t < 4; ++t) acc[t] = z;
  unsigned ldsA = (unsigned)(uintptr_t)As;
  unsigned ldsB = (unsigned)(uintptr_t)Bs;
  for (int kc = 0; kc < K; kc += 64) {
    unsigned long long gA = (unsigned long long)(uintptr_t)(A + row0 * (size_t)K + kc);
    unsigned long long gB = (unsigned long long)(uintptr_t)(Wt + (size_t)col0 * K + kc);
    // 64x64 bf16 tile = 512 x 16B segments; 128 threads x 4 segments each
    #pragma unroll
    for (int j = 0; j < 4; ++j) {
      int idx = tid + j * 128;            // 0..511
      int r = idx >> 3, c = idx & 7;      // row, 16B-column
      unsigned voff = (unsigned)((r * K + c * 8) * 2);   // global byte offset
      unsigned lo   = (unsigned)(idx * 16);              // LDS byte offset (row-major 64-wide)
      async_ld_b128(ldsA + lo, voff, gA);
      async_ld_b128(ldsB + lo, voff, gB);
    }
    wait_async0();
    __syncthreads();
    #pragma unroll
    for (int ks = 0; ks < 64; ks += 32) {
      v16bf af = load_a_frag(As, 64, wave * 16, ks, lane);
      #pragma unroll
      for (int t = 0; t < 4; ++t) {
        v16bf bfrag = load_b_frag_nmaj(Bs, 64, ks, t * 16, lane);
        acc[t] = wmma_bf16(af, bfrag, acc[t]);
      }
    }
    __syncthreads();
  }
}

// ---------- kernels ----------
__global__ void cvt_f32_bf16(const float* __restrict__ s, bf16* __restrict__ d, int n) {
  int i = blockIdx.x * 256 + threadIdx.x;
  if (i < n) d[i] = f2bf(s[i]);
}

// expand rel-pos bias to [NH, 64, 64] fp32
__global__ void bias_expand(const float* __restrict__ rpb, const int* __restrict__ rel,
                            float* __restrict__ biasNH) {
  int i = blockIdx.x * 256 + threadIdx.x;
  if (i >= 4096) return;
  int idx = rel[i];
  #pragma unroll
  for (int h = 0; h < NHEAD; ++h) biasNH[h * 4096 + i] = rpb[idx * NHEAD + h];
}

// LN1 + cyclic shift + window partition -> bf16 window-major activations
__global__ void ln_shift_win(const float* __restrict__ x, const float* __restrict__ g,
                             const float* __restrict__ b, bf16* __restrict__ win) {
  int wave = threadIdx.x >> 5, lane = threadIdx.x & 31;
  size_t tok = (size_t)blockIdx.x * 8 + wave;
  const float* px = x + tok * CDIM;
  float v[6], s = 0.f, ss = 0.f;
  #pragma unroll
  for (int j = 0; j < 6; ++j) { v[j] = px[lane + 32 * j]; s += v[j]; ss += v[j] * v[j]; }
  #pragma unroll
  for (int m = 16; m >= 1; m >>= 1) { s += __shfl_xor(s, m, 32); ss += __shfl_xor(ss, m, 32); }
  float mean = s * (1.0f / CDIM);
  float var = ss * (1.0f / CDIM) - mean * mean;
  float rstd = rsqrtf(var + 1e-5f);
  // token -> shifted window slot
  int bb = (int)(tok / 49152);
  int hw = (int)(tok % 49152);
  int h = hw / 384, w = hw % 384;
  int hs = h - 4; if (hs < 0) hs += 128;
  int ws = w - 4; if (ws < 0) ws += 384;
  int wi = (hs >> 3) * 48 + (ws >> 3);
  int t  = ((hs & 7) << 3) + (ws & 7);
  bf16* pd = win + (((size_t)bb * NWIN + wi) * NTOK + t) * CDIM;
  #pragma unroll
  for (int j = 0; j < 6; ++j) {
    int c = lane + 32 * j;
    pd[c] = f2bf((v[j] - mean) * rstd * g[c] + b[c]);
  }
}

// plain LN -> bf16 (token-major)
__global__ void ln_plain(const float* __restrict__ x, const float* __restrict__ g,
                         const float* __restrict__ b, bf16* __restrict__ out) {
  int wave = threadIdx.x >> 5, lane = threadIdx.x & 31;
  size_t tok = (size_t)blockIdx.x * 8 + wave;
  const float* px = x + tok * CDIM;
  float v[6], s = 0.f, ss = 0.f;
  #pragma unroll
  for (int j = 0; j < 6; ++j) { v[j] = px[lane + 32 * j]; s += v[j]; ss += v[j] * v[j]; }
  #pragma unroll
  for (int m = 16; m >= 1; m >>= 1) { s += __shfl_xor(s, m, 32); ss += __shfl_xor(ss, m, 32); }
  float mean = s * (1.0f / CDIM);
  float var = ss * (1.0f / CDIM) - mean * mean;
  float rstd = rsqrtf(var + 1e-5f);
  bf16* pd = out + tok * CDIM;
  #pragma unroll
  for (int j = 0; j < 6; ++j) {
    int c = lane + 32 * j;
    pd[c] = f2bf((v[j] - mean) * rstd * g[c] + b[c]);
  }
}

// QKV: [T,192] @ [576,192]^T + b -> bf16 [T,576]
__global__ void qkv_gemm(const bf16* __restrict__ A, const bf16* __restrict__ W,
                         const float* __restrict__ bias, bf16* __restrict__ out) {
  __shared__ bf16 As[64 * 64];
  __shared__ bf16 Bs[64 * 64];
  int tid = threadIdx.x, lane = tid & 31, wave = tid >> 5;
  size_t row0 = (size_t)blockIdx.x * 64;
  int col0 = blockIdx.y * 64;
  v8f acc[4];
  mm_mainloop(A, W, CDIM, row0, col0, As, Bs, acc, tid, lane, wave);
  int lhalf = lane >> 4, lcol = lane & 15;
  #pragma unroll
  for (int t = 0; t < 4; ++t) {
    union { v8f v; float e[8]; } au; au.v = acc[t];
    int col = col0 + t * 16 + lcol;
    float bv = bias[col];
    #pragma unroll
    for (int r = 0; r < 8; ++r) {
      size_t row = row0 + wave * 16 + r + (lhalf << 3);
      out[row * C3 + col] = f2bf(au.e[r] + bv);
    }
  }
}

// attention per (window, head)
__global__ void attn_win(const bf16* __restrict__ qkv, const float* __restrict__ biasNH,
                         const float* __restrict__ mask, bf16* __restrict__ aout) {
  __shared__ bf16 Qs[NTOK * HDIM];
  __shared__ bf16 Ks[NTOK * HDIM];
  __shared__ bf16 Vs[NTOK * HDIM];
  __shared__ float S[NTOK * NTOK];
  __shared__ bf16 P[NTOK * NTOK];
  int tid = threadIdx.x, lane = tid & 31, wave = tid >> 5;
  int head = blockIdx.x % NHEAD;
  size_t w = blockIdx.x / NHEAD;
  int wi = (int)(w % NWIN);

  // stage Q,K,V (each 64x32 bf16 = 256 x 16B) via async global->LDS copies
  {
    unsigned long long gbase = (unsigned long long)(uintptr_t)(qkv + w * (size_t)NTOK * C3);
    unsigned ldsq[3] = { (unsigned)(uintptr_t)Qs, (unsigned)(uintptr_t)Ks,
                         (unsigned)(uintptr_t)Vs };
    #pragma unroll
    for (int qi = 0; qi < 3; ++qi) {
      #pragma unroll
      for (int j = 0; j < 2; ++j) {
        int idx = tid + j * 128;          // 0..255
        int r = idx >> 2, c = idx & 3;    // row, 16B-column (row = 64B in LDS)
        unsigned voff = (unsigned)(r * (C3 * 2) + qi * 384 + head * 64 + c * 16);
        async_ld_b128(ldsq[qi] + (unsigned)(idx * 16), voff, gbase);
      }
    }
    wait_async0();
  }
  __syncthreads();

  // S = (Q K^T)*scale + bias + mask
  {
    v8f z = {};
    v8f acc[4] = { z, z, z, z };
    v16bf aq = load_a_frag(Qs, HDIM, wave * 16, 0, lane);
    #pragma unroll
    for (int t = 0; t < 4; ++t)
      acc[t] = wmma_bf16(aq, load_b_frag_nmaj(Ks, HDIM, 0, t * 16, lane), acc[t]);
    int lhalf = lane >> 4, lcol = lane & 15;
    const float* bh = biasNH + head * 4096;
    const float* mk = mask + (size_t)wi * 4096;
    #pragma unroll
    for (int t = 0; t < 4; ++t) {
      union { v8f v; float e[8]; } au; au.v = acc[t];
      #pragma unroll
      for (int r = 0; r < 8; ++r) {
        int row = wave * 16 + r + (lhalf << 3);
        int ij = row * NTOK + t * 16 + lcol;
        S[ij] = au.e[r] * SCALE_F + bh[ij] + mk[ij];
      }
    }
  }
  __syncthreads();

  // row softmax (fp32), P = softmax(S) in bf16
  if (tid < NTOK) {
    float mx = -1e30f;
    #pragma unroll 8
    for (int j = 0; j < NTOK; ++j) mx = fmaxf(mx, S[tid * NTOK + j]);
    float sum = 0.f;
    #pragma unroll 8
    for (int j = 0; j < NTOK; ++j) {
      float e = __expf(S[tid * NTOK + j] - mx);
      S[tid * NTOK + j] = e;
      sum += e;
    }
    float inv = 1.0f / sum;
    #pragma unroll 8
    for (int j = 0; j < NTOK; ++j) P[tid * NTOK + j] = f2bf(S[tid * NTOK + j] * inv);
  }
  __syncthreads();

  // O = P V  (64x32)
  v8f z2 = {};
  v8f acc2[2] = { z2, z2 };
  #pragma unroll
  for (int kc = 0; kc < NTOK; kc += 32) {
    v16bf ap = load_a_frag(P, NTOK, wave * 16, kc, lane);
    #pragma unroll
    for (int t = 0; t < 2; ++t)
      acc2[t] = wmma_bf16(ap, load_b_frag_kmaj(Vs, HDIM, kc, t * 16, lane), acc2[t]);
  }
  int lhalf = lane >> 4, lcol = lane & 15;
  #pragma unroll
  for (int t = 0; t < 2; ++t) {
    union { v8f v; float e[8]; } au; au.v = acc2[t];
    #pragma unroll
    for (int r = 0; r < 8; ++r) {
      int row = wave * 16 + r + (lhalf << 3);
      aout[(w * NTOK + row) * CDIM + head * HDIM + t * 16 + lcol] = f2bf(au.e[r]);
    }
  }
}

// proj GEMM + window-reverse + unshift + residual -> fp32 x1
__global__ void proj_gemm(const bf16* __restrict__ A, const bf16* __restrict__ W,
                          const float* __restrict__ bias, const float* __restrict__ xres,
                          float* __restrict__ x1) {
  __shared__ bf16 As[64 * 64];
  __shared__ bf16 Bs[64 * 64];
  int tid = threadIdx.x, lane = tid & 31, wave = tid >> 5;
  size_t row0 = (size_t)blockIdx.x * 64;
  int col0 = blockIdx.y * 64;
  v8f acc[4];
  mm_mainloop(A, W, CDIM, row0, col0, As, Bs, acc, tid, lane, wave);
  int lhalf = lane >> 4, lcol = lane & 15;
  #pragma unroll
  for (int r = 0; r < 8; ++r) {
    size_t wrow = row0 + wave * 16 + r + (lhalf << 3);
    int t64 = (int)(wrow & 63);
    size_t wn = wrow >> 6;
    int bb = (int)(wn / NWIN);
    int wi = (int)(wn % NWIN);
    int hs = (wi / 48) * 8 + (t64 >> 3);
    int ws = (wi % 48) * 8 + (t64 & 7);
    int h = hs + 4; if (h >= 128) h -= 128;
    int ww = ws + 4; if (ww >= 384) ww -= 384;
    size_t tok = ((size_t)bb * 128 + h) * 384 + ww;
    #pragma unroll
    for (int t = 0; t < 4; ++t) {
      union { v8f v; float e[8]; } au; au.v = acc[t];
      int col = col0 + t * 16 + lcol;
      x1[tok * CDIM + col] = xres[tok * CDIM + col] + au.e[r] + bias[col];
    }
  }
}

// fc1 GEMM + exact GELU -> bf16 [T, 768]
__global__ void fc1_gemm(const bf16* __restrict__ A, const bf16* __restrict__ W,
                         const float* __restrict__ bias, bf16* __restrict__ out) {
  __shared__ bf16 As[64 * 64];
  __shared__ bf16 Bs[64 * 64];
  int tid = threadIdx.x, lane = tid & 31, wave = tid >> 5;
  size_t row0 = (size_t)blockIdx.x * 64;
  int col0 = blockIdx.y * 64;
  v8f acc[4];
  mm_mainloop(A, W, CDIM, row0, col0, As, Bs, acc, tid, lane, wave);
  int lhalf = lane >> 4, lcol = lane & 15;
  #pragma unroll
  for (int t = 0; t < 4; ++t) {
    union { v8f v; float e[8]; } au; au.v = acc[t];
    int col = col0 + t * 16 + lcol;
    float bv = bias[col];
    #pragma unroll
    for (int r = 0; r < 8; ++r) {
      size_t row = row0 + wave * 16 + r + (lhalf << 3);
      float v = au.e[r] + bv;
      float gv = 0.5f * v * (1.0f + erff(v * 0.70710678118654752f));
      out[row * MLPH + col] = f2bf(gv);
    }
  }
}

// fc2 GEMM + bias + residual -> fp32 out
__global__ void fc2_gemm(const bf16* __restrict__ A, const bf16* __restrict__ W,
                         const float* __restrict__ bias, const float* __restrict__ x1,
                         float* __restrict__ out) {
  __shared__ bf16 As[64 * 64];
  __shared__ bf16 Bs[64 * 64];
  int tid = threadIdx.x, lane = tid & 31, wave = tid >> 5;
  size_t row0 = (size_t)blockIdx.x * 64;
  int col0 = blockIdx.y * 64;
  v8f acc[4];
  mm_mainloop(A, W, MLPH, row0, col0, As, Bs, acc, tid, lane, wave);
  int lhalf = lane >> 4, lcol = lane & 15;
  #pragma unroll
  for (int t = 0; t < 4; ++t) {
    union { v8f v; float e[8]; } au; au.v = acc[t];
    int col = col0 + t * 16 + lcol;
    float bv = bias[col];
    #pragma unroll
    for (int r = 0; r < 8; ++r) {
      size_t row = row0 + wave * 16 + r + (lhalf << 3);
      out[row * CDIM + col] = x1[row * CDIM + col] + au.e[r] + bv;
    }
  }
}

// ---------- launch ----------
static inline size_t alignup(size_t v) { return (v + 255) & ~(size_t)255; }

extern "C" void kernel_launch(void* const* d_in, const int* in_sizes, int n_in,
                              void* d_out, int out_size, void* d_ws, size_t ws_size,
                              hipStream_t stream) {
  (void)in_sizes; (void)n_in; (void)out_size; (void)ws_size;
  const float* x      = (const float*)d_in[0];
  const float* qkv_w  = (const float*)d_in[1];
  const float* qkv_b  = (const float*)d_in[2];
  const float* rpb    = (const float*)d_in[3];
  const float* proj_w = (const float*)d_in[4];
  const float* proj_b = (const float*)d_in[5];
  const float* g1     = (const float*)d_in[6];
  const float* b1     = (const float*)d_in[7];
  const float* g2     = (const float*)d_in[8];
  const float* b2     = (const float*)d_in[9];
  const float* fc1_w  = (const float*)d_in[10];
  const float* fc1_b  = (const float*)d_in[11];
  const float* fc2_w  = (const float*)d_in[12];
  const float* fc2_b  = (const float*)d_in[13];
  const int*   rel    = (const int*)d_in[14];
  const float* mask   = (const float*)d_in[15];
  float* out = (float*)d_out;

  char* ws = (char*)d_ws;
  size_t o = 0;
  bf16*  win    = (bf16*)(ws + o);  o += alignup(TOKENS * CDIM * 2);
  bf16*  qkv    = (bf16*)(ws + o);  o += alignup(TOKENS * C3 * 2);
  bf16*  aout   = (bf16*)(ws + o);  o += alignup(TOKENS * CDIM * 2);
  float* x1     = (float*)(ws + o); o += alignup(TOKENS * CDIM * 4);
  bf16*  h2     = (bf16*)(ws + o);  o += alignup(TOKENS * CDIM * 2);
  bf16*  a1     = (bf16*)(ws + o);  o += alignup(TOKENS * MLPH * 2);
  bf16*  wqkv   = (bf16*)(ws + o);  o += alignup((size_t)C3 * CDIM * 2);
  bf16*  wproj  = (bf16*)(ws + o);  o += alignup((size_t)CDIM * CDIM * 2);
  bf16*  wfc1   = (bf16*)(ws + o);  o += alignup((size_t)MLPH * CDIM * 2);
  bf16*  wfc2   = (bf16*)(ws + o);  o += alignup((size_t)CDIM * MLPH * 2);
  float* biasNH = (float*)(ws + o); o += alignup((size_t)NHEAD * 4096 * 4);

  // weight conversion to bf16
  cvt_f32_bf16<<<(C3 * CDIM + 255) / 256, 256, 0, stream>>>(qkv_w, wqkv, C3 * CDIM);
  cvt_f32_bf16<<<(CDIM * CDIM + 255) / 256, 256, 0, stream>>>(proj_w, wproj, CDIM * CDIM);
  cvt_f32_bf16<<<(MLPH * CDIM + 255) / 256, 256, 0, stream>>>(fc1_w, wfc1, MLPH * CDIM);
  cvt_f32_bf16<<<(CDIM * MLPH + 255) / 256, 256, 0, stream>>>(fc2_w, wfc2, CDIM * MLPH);
  bias_expand<<<16, 256, 0, stream>>>(rpb, rel, biasNH);

  // LN1 + shift + window partition
  ln_shift_win<<<(unsigned)(TOKENS / 8), 256, 0, stream>>>(x, g1, b1, win);

  // QKV projection
  qkv_gemm<<<dim3((unsigned)(TOKENS / 64), C3 / 64), 128, 0, stream>>>(win, wqkv, qkv_b, qkv);

  // windowed attention
  attn_win<<<(unsigned)(TOKENS / NTOK * NHEAD), 128, 0, stream>>>(qkv, biasNH, mask, aout);

  // proj + window reverse + unshift + residual
  proj_gemm<<<dim3((unsigned)(TOKENS / 64), CDIM / 64), 128, 0, stream>>>(aout, wproj, proj_b, x, x1);

  // LN2
  ln_plain<<<(unsigned)(TOKENS / 8), 256, 0, stream>>>(x1, g2, b2, h2);

  // MLP
  fc1_gemm<<<dim3((unsigned)(TOKENS / 64), MLPH / 64), 128, 0, stream>>>(h2, wfc1, fc1_b, a1);
  fc2_gemm<<<dim3((unsigned)(TOKENS / 64), CDIM / 64), 128, 0, stream>>>(a1, wfc2, fc2_b, x1, out);
}
